// Encoder_1975684956952
// MI455X (gfx1250) — matrix operand order repaired and checked by reading, not compiled, not run on hardware
//
#include <hip/hip_runtime.h>
#include <hip/hip_bf16.h>

// GRU encoder for MI455X (gfx1250, wave32, WMMA).
// B=64, S=512, E=512, H=1024.
//
//  - bf16 WMMA (v_wmma_f32_16x16x32_bf16), f32 accumulate.
//  - Pre-transposed weights WT[g][n][k] so each lane's B-fragment is a
//    contiguous 32B run (B order used here, same as A: lanes0-15 K=lo16, lanes16-31 K=hi16,
//    column = lane&15).
//  - xproj: block = 128 rows x 64 cols; B tile staged in LDS (padded,
//    bank-conflict-free), 8 waves share it; each wave = 16x64 strip with
//    4 independent WMMA accumulator chains; all 4 B fragments preloaded per
//    k-chunk so s_wait_dscnt counts down behind the WMMA stream.
//  - scan: per-step gates GEMM then candidate GEMM, 1 tile/wave for max
//    parallelism (latency-critical), dual accumulator chains per wave.

#define B_   64
#define S_   512
#define E_   512
#define H_   1024
#define EH_  (E_ + H_)

typedef __attribute__((ext_vector_type(16))) __bf16 v16bf;
typedef __attribute__((ext_vector_type(8)))  __bf16 v8bf;
typedef __attribute__((ext_vector_type(8)))  float  v8f;

__device__ __forceinline__ float sigmoidf_(float x) {
    return 1.0f / (1.0f + __expf(-x));
}

__device__ __forceinline__ v16bf make_a_frag(const __bf16* __restrict__ arow,
                                             int k, int hi) {
    // A layout: lane<16 holds K {k..k+15}; lane>=16 holds K {k+16..k+31}.
    v8bf a0 = *(const v8bf*)(arow + k + hi * 16);
    v8bf a1 = *(const v8bf*)(arow + k + hi * 16 + 8);
    v16bf a;
    #pragma unroll
    for (int i = 0; i < 8; ++i) { a[i] = a0[i]; a[8 + i] = a1[i]; }
    return a;
}

__device__ __forceinline__ v16bf lds_b_frag(const __bf16* bp) {
    v8bf b0 = *(const v8bf*)(bp);
    v8bf b1 = *(const v8bf*)(bp + 8);
    v16bf b;
    #pragma unroll
    for (int i = 0; i < 8; ++i) { b[i] = b0[i]; b[8 + i] = b1[i]; }
    return b;
}

// One 16x16 tile with two independent accumulator chains (halves the exposed
// WMMA->WMMA dependency depth). A row-major (lda), BT row-major by output
// column (ldb), K multiple of 64.
__device__ __forceinline__ v8f wmma_row_dual(const __bf16* __restrict__ A, int lda,
                                             const __bf16* __restrict__ BT, int ldb,
                                             int K) {
    const int lane = threadIdx.x & 31;
    const int r    = lane & 15;
    const int hi   = lane >> 4;
    const __bf16* arow = A  + (size_t)r * lda;
    const __bf16* brow = BT + (size_t)r * ldb;
    v8f acc0 = {}, acc1 = {};
    #pragma unroll 2
    for (int k = 0; k < K; k += 64) {
        v16bf a0 = make_a_frag(arow, k, hi);
        v16bf b0 = *(const v16bf*)(brow + k + hi * 16);
        v16bf a1 = make_a_frag(arow, k + 32, hi);
        v16bf b1 = *(const v16bf*)(brow + k + 32 + hi * 16);
        __builtin_prefetch(brow + k + 256, 0, 1);           // global_prefetch_b8
        acc0 = __builtin_amdgcn_wmma_f32_16x16x32_bf16(
            false, a0, false, b0, (short)0, acc0, false, false);
        acc1 = __builtin_amdgcn_wmma_f32_16x16x32_bf16(
            false, a1, false, b1, (short)0, acc1, false, false);
    }
    v8f acc;
    #pragma unroll
    for (int i = 0; i < 8; ++i) acc[i] = acc0[i] + acc1[i];
    return acc;
}

// ---------------- prep kernels ----------------

// embedding [B,S,E] f32 -> X_bf [S,B,E] bf16
__global__ void cast_embed_kernel(const float* __restrict__ emb,
                                  __bf16* __restrict__ xbf) {
    const long total = (long)B_ * S_ * E_;
    for (long i = (long)blockIdx.x * blockDim.x + threadIdx.x; i < total;
         i += (long)gridDim.x * blockDim.x) {
        long e = i % E_;
        long s = (i / E_) % S_;
        long b = i / ((long)E_ * S_);
        xbf[(s * B_ + b) * (long)E_ + e] = (__bf16)emb[(b * S_ + s) * (long)E_ + e];
    }
}

// W[g] [EH,H] f32 -> WT[g] [H,EH] bf16 (transpose + cast)
__global__ void cast_w_kernel(const float* __restrict__ Wz,
                              const float* __restrict__ Wr,
                              const float* __restrict__ Wm,
                              __bf16* __restrict__ WT) {
    const long per = (long)H_ * EH_;
    const long total = 3 * per;
    for (long i = (long)blockIdx.x * blockDim.x + threadIdx.x; i < total;
         i += (long)gridDim.x * blockDim.x) {
        int g = (int)(i / per);
        long j = i % per;
        long n = j / EH_;
        long k = j % EH_;
        const float* W = (g == 0) ? Wz : (g == 1) ? Wr : Wm;
        WT[i] = (__bf16)W[k * H_ + n];
    }
}

__global__ void init_h_kernel(float* __restrict__ h, __bf16* __restrict__ hbf) {
    int i = blockIdx.x * blockDim.x + threadIdx.x;
    if (i < B_ * H_) { h[i] = 0.0f; hbf[i] = (__bf16)0.0f; }
}

// ---------------- time-parallel input projection ----------------
// Block: 128 rows x 64 cols. B tile staged in LDS in two K phases of 256.
// Each wave: one 16-row tile x all 64 cols (4 accumulator chains).

#define KC_   256
#define PADK_ (KC_ + 8)     // 264 elements -> 528B row stride, conflict-free

__global__ __launch_bounds__(256)
void xproj_kernel(const __bf16* __restrict__ xbf,
                  const __bf16* __restrict__ WT,
                  const float* __restrict__ bz,
                  const float* __restrict__ br,
                  const float* __restrict__ bm,
                  float* __restrict__ Xproj) {
    __shared__ __bf16 Bsm[64 * PADK_];                  // 33,792 bytes

    const int gate  = blockIdx.z;                       // 0=z 1=r 2=m
    const int nquad = blockIdx.y;                       // 16 quads of 64 cols
    const long msup = blockIdx.x;                       // 256 tiles of 128 rows
    const int tid   = threadIdx.x;
    const int wave  = tid >> 5;
    const int lane  = tid & 31;
    const int r     = lane & 15;
    const int hi    = lane >> 4;

    const __bf16* Wg = WT + (size_t)gate * H_ * EH_ + (size_t)nquad * 64 * EH_;
    const long row0 = msup * 128 + wave * 16;
    const __bf16* arow = xbf + (row0 + r) * (long)E_;

    v8f acc[4] = {};
    for (int kp = 0; kp < E_; kp += KC_) {
        // cooperative stage: 64 cols x 256 K of bf16 weights -> LDS
        for (int idx = tid; idx < 64 * (KC_ / 8); idx += 256) {
            int col = idx >> 5;                          // KC_/8 == 32 chunks
            int kb  = idx & 31;
            *(v8bf*)&Bsm[col * PADK_ + kb * 8] =
                *(const v8bf*)(Wg + (size_t)col * EH_ + kp + kb * 8);
        }
        __syncthreads();

        for (int k = 0; k < KC_; k += 32) {
            v16bf a = make_a_frag(arow, kp + k, hi);
            // preload all 4 B fragments so dscnt waits overlap the WMMA stream
            v16bf b0 = lds_b_frag(&Bsm[(0 * 16 + r) * PADK_ + k + hi * 16]);
            v16bf b1 = lds_b_frag(&Bsm[(1 * 16 + r) * PADK_ + k + hi * 16]);
            v16bf b2 = lds_b_frag(&Bsm[(2 * 16 + r) * PADK_ + k + hi * 16]);
            v16bf b3 = lds_b_frag(&Bsm[(3 * 16 + r) * PADK_ + k + hi * 16]);
            acc[0] = __builtin_amdgcn_wmma_f32_16x16x32_bf16(
                false, a, false, b0, (short)0, acc[0], false, false);
            acc[1] = __builtin_amdgcn_wmma_f32_16x16x32_bf16(
                false, a, false, b1, (short)0, acc[1], false, false);
            acc[2] = __builtin_amdgcn_wmma_f32_16x16x32_bf16(
                false, a, false, b2, (short)0, acc[2], false, false);
            acc[3] = __builtin_amdgcn_wmma_f32_16x16x32_bf16(
                false, a, false, b3, (short)0, acc[3], false, false);
        }
        __syncthreads();
    }

    const float* bias = (gate == 0) ? bz : (gate == 1) ? br : bm;
    float* out = Xproj + (long)gate * S_ * B_ * H_;
    #pragma unroll
    for (int t = 0; t < 4; ++t) {
        const int col = nquad * 64 + t * 16 + r;
        const float bv = bias[col];
        #pragma unroll
        for (int i = 0; i < 8; ++i) {
            long row = row0 + i + hi * 8;
            out[row * (long)H_ + col] = acc[t][i] + bv;
        }
    }
}

// ---------------- per-step kernels ----------------
// Stage 1: z = sigmoid(Xz[s] + h @ Wz_h), r = sigmoid(Xr[s] + h @ Wr_h)
__global__ __launch_bounds__(256)
void gates_kernel(const __bf16* __restrict__ hbf,
                  const float* __restrict__ h,
                  const __bf16* __restrict__ WT,
                  const float* __restrict__ Xproj,
                  float* __restrict__ zbuf,
                  __bf16* __restrict__ rhbf,
                  int s) {
    const int wave = threadIdx.x >> 5;
    const int lane = threadIdx.x & 31;
    const int tile = blockIdx.x * 8 + wave;             // 512 waves total
    const int gate = tile >> 8;                         // 0=z, 1=r
    const int t    = tile & 255;
    const int tileM = t >> 6;                           // 4 row tiles (B=64)
    const int tileN = t & 63;                           // 64 col tiles

    const __bf16* A  = hbf + tileM * 16 * H_;
    const __bf16* BT = WT + (long)gate * H_ * EH_ + (long)tileN * 16 * EH_ + E_;
    v8f acc = wmma_row_dual(A, H_, BT, EH_, H_);

    const float* Xg = Xproj + (long)gate * S_ * B_ * H_ + (long)s * B_ * H_;
    const int r  = lane & 15;
    const int hi = lane >> 4;
    const int col = tileN * 16 + r;
    #pragma unroll
    for (int i = 0; i < 8; ++i) {
        int row = tileM * 16 + i + hi * 8;
        float g = sigmoidf_(acc[i] + Xg[row * H_ + col]);
        if (gate == 0) {
            zbuf[row * H_ + col] = g;
        } else {
            rhbf[row * H_ + col] = (__bf16)(g * h[row * H_ + col]);
        }
    }
}

// Stage 2: h1 = tanh(Xm[s] + (r*h) @ Wm_h); h = (1-z)*h + z*h1; out[b][s][:]
__global__ __launch_bounds__(256)
void cand_kernel(const __bf16* __restrict__ rhbf,
                 const __bf16* __restrict__ WT,
                 const float* __restrict__ Xproj,
                 const float* __restrict__ zbuf,
                 float* __restrict__ h,
                 __bf16* __restrict__ hbf,
                 float* __restrict__ out,
                 int s) {
    const int wave = threadIdx.x >> 5;
    const int lane = threadIdx.x & 31;
    const int tile = blockIdx.x * 8 + wave;             // 256 waves total
    const int tileM = tile >> 6;
    const int tileN = tile & 63;

    const __bf16* A  = rhbf + tileM * 16 * H_;
    const __bf16* BT = WT + 2L * H_ * EH_ + (long)tileN * 16 * EH_ + E_;
    v8f acc = wmma_row_dual(A, H_, BT, EH_, H_);

    const float* Xm = Xproj + 2L * S_ * B_ * H_ + (long)s * B_ * H_;
    const int r  = lane & 15;
    const int hi = lane >> 4;
    const int col = tileN * 16 + r;
    #pragma unroll
    for (int i = 0; i < 8; ++i) {
        int row = tileM * 16 + i + hi * 8;              // row == batch index b
        float h1 = tanhf(acc[i] + Xm[row * H_ + col]);
        float z  = zbuf[row * H_ + col];
        float hn = (1.0f - z) * h[row * H_ + col] + z * h1;
        h[row * H_ + col]   = hn;
        hbf[row * H_ + col] = (__bf16)hn;
        out[((long)row * S_ + s) * (long)H_ + col] = hn;   // [B,S,H]
    }
}

// ---------------- host launcher ----------------

extern "C" void kernel_launch(void* const* d_in, const int* in_sizes, int n_in,
                              void* d_out, int out_size, void* d_ws, size_t ws_size,
                              hipStream_t stream) {
    (void)in_sizes; (void)n_in; (void)out_size; (void)ws_size;

    const float* emb = (const float*)d_in[0];
    const float* Wz  = (const float*)d_in[1];
    const float* bz  = (const float*)d_in[2];
    const float* Wr  = (const float*)d_in[3];
    const float* br  = (const float*)d_in[4];
    const float* Wm  = (const float*)d_in[5];
    const float* bm  = (const float*)d_in[6];
    float* out = (float*)d_out;

    // workspace carve-up (256B aligned)
    char* ws = (char*)d_ws;
    size_t off = 0;
    auto alloc = [&](size_t bytes) {
        void* p = ws + off;
        off += (bytes + 255) & ~(size_t)255;
        return p;
    };
    __bf16* xbf   = (__bf16*)alloc((size_t)S_ * B_ * E_ * sizeof(__bf16));   // 33.5MB
    __bf16* WT    = (__bf16*)alloc((size_t)3 * H_ * EH_ * sizeof(__bf16));   // 9.4MB
    float*  Xproj = (float*) alloc((size_t)3 * S_ * B_ * H_ * sizeof(float));// 402MB
    float*  h     = (float*) alloc((size_t)B_ * H_ * sizeof(float));
    __bf16* hbf   = (__bf16*)alloc((size_t)B_ * H_ * sizeof(__bf16));
    float*  zbuf  = (float*) alloc((size_t)B_ * H_ * sizeof(float));
    __bf16* rhbf  = (__bf16*)alloc((size_t)B_ * H_ * sizeof(__bf16));

    // prep
    cast_embed_kernel<<<4096, 256, 0, stream>>>(emb, xbf);
    cast_w_kernel<<<4096, 256, 0, stream>>>(Wz, Wr, Wm, WT);
    init_h_kernel<<<(B_ * H_ + 255) / 256, 256, 0, stream>>>(h, hbf);

    // time-parallel input projections:
    // grid = (32768/128 row-supertiles, 1024/64 col-quads, 3 gates)
    {
        dim3 grid(256, 16, 3);
        xproj_kernel<<<grid, 256, 0, stream>>>(xbf, WT, bz, br, bm, Xproj);
    }

    // sequential scan
    for (int s = 0; s < S_; ++s) {
        gates_kernel<<<64, 256, 0, stream>>>(hbf, h, WT, Xproj, zbuf, rhbf, s);
        cand_kernel<<<32, 256, 0, stream>>>(rhbf, WT, Xproj, zbuf, h, hbf, out, s);
    }
}
